// GAT_22170621182206
// MI455X (gfx1250) — matrix operand order repaired and checked by reading, not compiled, not run on hardware
//
#include <hip/hip_runtime.h>
#include <hip/hip_bf16.h>

typedef __attribute__((ext_vector_type(16))) __bf16 v16bf;
typedef __attribute__((ext_vector_type(8)))  float  v8f;

#define NEG_SLOPE 0.2f

__device__ __forceinline__ unsigned f2ord(float f) {
    unsigned u = __float_as_uint(f);
    return (u & 0x80000000u) ? ~u : (u | 0x80000000u);
}
__device__ __forceinline__ float ord2f(unsigned u) {
    return __uint_as_float((u & 0x80000000u) ? (u & 0x7FFFFFFFu) : ~u);
}
__device__ __forceinline__ float lrelu(float x) { return x > 0.f ? x : NEG_SLOPE * x; }

// ---------------------------------------------------------------------------
// Dense GEMM  H[N,Dout] = X[N,K] @ W[K,Dout],  Dout = TN*16.
// One wave per 16-row strip; TN WMMAs per K-chunk share one A load.
// A rows are CLAMPED (not zeroed) out of range: the corresponding D rows are
// masked at store time, so garbage there is harmless -> no divergent loads.
// ---------------------------------------------------------------------------
template <int TN>
__global__ void gat_gemm_wmma(const float* __restrict__ X,
                              const float* __restrict__ Wm,
                              float* __restrict__ H,
                              int N, int K)
{
    const int Dout  = TN * 16;
    const int lane  = threadIdx.x & 31;
    const int wave  = blockIdx.x * (blockDim.x >> 5) + (threadIdx.x >> 5);
    const int mtiles = (N + 15) >> 4;
    if (wave >= mtiles) return;                   // wave-uniform: EXEC stays full

    const int mtile = wave;
    const int half  = lane >> 4;                  // 0: lanes 0-15, 1: lanes 16-31
    const int mr    = lane & 15;
    int arow = mtile * 16 + mr;
    if (arow >= N) arow = N - 1;                  // clamp, see note above
    const float* __restrict__ xp = X + (size_t)arow * K;

    v8f cacc[TN];
#pragma unroll
    for (int j = 0; j < TN; ++j)
        cacc[j] = (v8f){0.f, 0.f, 0.f, 0.f, 0.f, 0.f, 0.f, 0.f};

    for (int k0 = 0; k0 < K; k0 += 32) {
        // A 16x32 bf16, this code's k order (B matches): lane half 0 holds K 0..15, half 1: 16..31.
        // Each run of 8 floats is contiguous -> two b128 loads per run.
        const float4* __restrict__ ap0 = (const float4*)(xp + k0 + half * 16);
        const float4* __restrict__ ap1 = (const float4*)(xp + k0 + half * 16 + 8);
        const float4 a0 = ap0[0], a1 = ap0[1];
        const float4 a2 = ap1[0], a3 = ap1[1];
        v16bf a;
        a[0]=(__bf16)a0.x; a[1]=(__bf16)a0.y; a[2]=(__bf16)a0.z; a[3]=(__bf16)a0.w;
        a[4]=(__bf16)a1.x; a[5]=(__bf16)a1.y; a[6]=(__bf16)a1.z; a[7]=(__bf16)a1.w;
        a[8]=(__bf16)a2.x; a[9]=(__bf16)a2.y; a[10]=(__bf16)a2.z; a[11]=(__bf16)a2.w;
        a[12]=(__bf16)a3.x; a[13]=(__bf16)a3.y; a[14]=(__bf16)a3.z; a[15]=(__bf16)a3.w;

        // B 32x16 bf16, same k order as A: col = lane%16, half 0 holds K 0..15, half 1 K 16..31.
        const float* __restrict__ wp = Wm + (size_t)(k0 + half * 16) * Dout + mr;
#pragma unroll
        for (int j = 0; j < TN; ++j) {
            v16bf b;
#pragma unroll
            for (int i = 0; i < 16; ++i)
                b[i] = (__bf16)wp[(size_t)i * Dout + j * 16];
            cacc[j] = __builtin_amdgcn_wmma_f32_16x16x32_bf16(false, a, false, b,
                                                              (short)0, cacc[j],
                                                              false, false);
        }
    }
    // D layout: VGPR r -> row M = r + 8*half, col = lane%16 (+ 16*j)
#pragma unroll
    for (int j = 0; j < TN; ++j) {
#pragma unroll
        for (int r = 0; r < 8; ++r) {
            const int m = mtile * 16 + half * 8 + r;
            if (m < N) H[(size_t)m * Dout + j * 16 + mr] = cacc[j][r];
        }
    }
}

// ---------------------------------------------------------------------------
// Per-(node,head) attention logits; also resets segment max / denom buffers.
// ---------------------------------------------------------------------------
__global__ void gat_attn_prep(const float* __restrict__ H,
                              const float* __restrict__ asrc,
                              const float* __restrict__ adst,
                              float* __restrict__ alsrc,
                              float* __restrict__ aldst,
                              unsigned* __restrict__ maxb,
                              float* __restrict__ denom,
                              int N, int Hh, int C)
{
    long long t = blockIdx.x * (long long)blockDim.x + threadIdx.x;
    if (t >= (long long)N * Hh) return;
    const int n = (int)(t / Hh);
    const int h = (int)(t % Hh);
    const float* __restrict__ hp = H + (size_t)n * Hh * C + (size_t)h * C;
    float s = 0.f, d = 0.f;
    for (int c = 0; c < C; ++c) {
        const float v = hp[c];
        s += v * asrc[h * C + c];
        d += v * adst[h * C + c];
    }
    alsrc[t] = s;
    aldst[t] = d;
    maxb[t]  = 0u;     // 0 < f2ord(x) for every real x; self-loop guarantees an update
    denom[t] = 0.f;
}

__global__ void gat_zero(float* __restrict__ p, long long n)
{
    long long t = blockIdx.x * (long long)blockDim.x + threadIdx.x;
    if (t < n) p[t] = 0.f;
}

__device__ __forceinline__ void edge_src_dst(const long long* __restrict__ ei,
                                             long long e, int E, int& src, int& dst)
{
    if (e < E) { src = (int)ei[e]; dst = (int)ei[(long long)E + e]; }
    else       { src = dst = (int)(e - E); }      // appended self-loops
}

// ---------------------------------------------------------------------------
// Pass 1: segment max of leaky-relu logits (float atomicMax via ordered uint)
// ---------------------------------------------------------------------------
__global__ void gat_edge_max(const long long* __restrict__ ei,
                             const float* __restrict__ alsrc,
                             const float* __restrict__ aldst,
                             unsigned* __restrict__ maxb,
                             int E, int N, int Hh)
{
    long long t = blockIdx.x * (long long)blockDim.x + threadIdx.x;
    const long long tot = ((long long)E + N) * Hh;
    if (t >= tot) return;
    const int h = (int)(t % Hh);
    const long long e = t / Hh;
    int src, dst;
    edge_src_dst(ei, e, E, src, dst);
    const float ev = lrelu(alsrc[(size_t)src * Hh + h] + aldst[(size_t)dst * Hh + h]);
    atomicMax(maxb + (size_t)dst * Hh + h, f2ord(ev));
}

// ---------------------------------------------------------------------------
// Pass 2: segment sum of exp(e - max)
// ---------------------------------------------------------------------------
__global__ void gat_edge_expsum(const long long* __restrict__ ei,
                                const float* __restrict__ alsrc,
                                const float* __restrict__ aldst,
                                const unsigned* __restrict__ maxb,
                                float* __restrict__ denom,
                                int E, int N, int Hh)
{
    long long t = blockIdx.x * (long long)blockDim.x + threadIdx.x;
    const long long tot = ((long long)E + N) * Hh;
    if (t >= tot) return;
    const int h = (int)(t % Hh);
    const long long e = t / Hh;
    int src, dst;
    edge_src_dst(ei, e, E, src, dst);
    const float ev = lrelu(alsrc[(size_t)src * Hh + h] + aldst[(size_t)dst * Hh + h]);
    const float m  = ord2f(maxb[(size_t)dst * Hh + h]);
    atomicAdd(denom + (size_t)dst * Hh + h, __expf(ev - m));
}

// ---------------------------------------------------------------------------
// Pass 3: acc[dst] += alpha * h[src]   (one thread per edge-channel)
// ---------------------------------------------------------------------------
__global__ void gat_edge_agg(const long long* __restrict__ ei,
                             const float* __restrict__ alsrc,
                             const float* __restrict__ aldst,
                             const unsigned* __restrict__ maxb,
                             const float* __restrict__ denom,
                             const float* __restrict__ H,
                             float* __restrict__ acc,
                             int E, int N, int Hh, int C)
{
    const int D = Hh * C;
    long long t = blockIdx.x * (long long)blockDim.x + threadIdx.x;
    const long long tot = ((long long)E + N) * D;
    if (t >= tot) return;
    const int c = (int)(t % D);
    const long long e = t / D;
    const int h = c / C;
    int src, dst;
    edge_src_dst(ei, e, E, src, dst);
    const float ev = lrelu(alsrc[(size_t)src * Hh + h] + aldst[(size_t)dst * Hh + h]);
    const float m  = ord2f(maxb[(size_t)dst * Hh + h]);
    const float alpha = __expf(ev - m) / denom[(size_t)dst * Hh + h];
    atomicAdd(acc + (size_t)dst * D + c, H[(size_t)src * D + c] * alpha);
}

__global__ void gat_finalize(const float* __restrict__ acc,
                             const float* __restrict__ bias,
                             float* __restrict__ out,
                             long long n, int D, int doRelu)
{
    long long t = blockIdx.x * (long long)blockDim.x + threadIdx.x;
    if (t >= n) return;
    const int c = (int)(t % D);
    float v = acc[t] + bias[c];
    if (doRelu) v = fmaxf(v, 0.f);
    out[t] = v;
}

// ---------------------------------------------------------------------------
// Host-side orchestration
// ---------------------------------------------------------------------------
static inline unsigned blocks_for(long long n) { return (unsigned)((n + 255) / 256); }

static void run_layer(const float* Xin, const float* Wm,
                      const float* asrc, const float* adst, const float* bias,
                      int K, int Hh, int C, int doRelu,
                      const long long* ei, int N, int E,
                      float* Hb, float* acc, float* alsrc, float* aldst,
                      unsigned* maxb, float* denom, float* outp,
                      hipStream_t stream)
{
    const int D = Hh * C;
    const long long waves = (long long)((N + 15) / 16);
    if (D == 64)
        gat_gemm_wmma<4><<<blocks_for(waves * 32), 256, 0, stream>>>(Xin, Wm, Hb, N, K);
    else
        gat_gemm_wmma<2><<<blocks_for(waves * 32), 256, 0, stream>>>(Xin, Wm, Hb, N, K);

    const long long nh = (long long)N * Hh;
    gat_attn_prep<<<blocks_for(nh), 256, 0, stream>>>(Hb, asrc, adst, alsrc, aldst,
                                                      maxb, denom, N, Hh, C);
    const long long nd = (long long)N * D;
    gat_zero<<<blocks_for(nd), 256, 0, stream>>>(acc, nd);

    const long long ET = (long long)E + N;
    gat_edge_max   <<<blocks_for(ET * Hh), 256, 0, stream>>>(ei, alsrc, aldst, maxb, E, N, Hh);
    gat_edge_expsum<<<blocks_for(ET * Hh), 256, 0, stream>>>(ei, alsrc, aldst, maxb, denom, E, N, Hh);
    gat_edge_agg   <<<blocks_for(ET * D),  256, 0, stream>>>(ei, alsrc, aldst, maxb, denom,
                                                             Hb, acc, E, N, Hh, C);
    gat_finalize<<<blocks_for(nd), 256, 0, stream>>>(acc, bias, outp, nd, D, doRelu);
}

extern "C" void kernel_launch(void* const* d_in, const int* in_sizes, int n_in,
                              void* d_out, int out_size, void* d_ws, size_t ws_size,
                              hipStream_t stream)
{
    const float*     x      = (const float*)d_in[0];
    const long long* ei     = (const long long*)d_in[1];   // int64 edge_index [2,E] flat
    const float*     W1     = (const float*)d_in[2];
    const float*     asrc1  = (const float*)d_in[3];
    const float*     adst1  = (const float*)d_in[4];
    const float*     b1     = (const float*)d_in[5];
    const float*     W2     = (const float*)d_in[6];
    const float*     asrc2  = (const float*)d_in[7];
    const float*     adst2  = (const float*)d_in[8];
    const float*     b2     = (const float*)d_in[9];
    const float*     W3     = (const float*)d_in[10];
    const float*     asrc3  = (const float*)d_in[11];
    const float*     adst3  = (const float*)d_in[12];
    const float*     b3     = (const float*)d_in[13];

    const int IN_C = 128, HID = 16, HEADS = 4, OUT_C = 32;
    const int N = in_sizes[0] / IN_C;
    const int E = in_sizes[1] / 2;
    const int D = HEADS * HID;                    // 64

    // Workspace carve-up (all fp32 unless noted). ~83 MB total; fits 192 MB L2.
    float*    xbuf  = (float*)d_ws;               // [N, 64]  layer outputs -> next input
    float*    hbuf  = xbuf  + (size_t)N * D;      // [N, 64]  h = x @ W
    float*    acc   = hbuf  + (size_t)N * D;      // [N, 64]  segment-sum accumulator
    float*    alsrc = acc   + (size_t)N * D;      // [N, 4]
    float*    aldst = alsrc + (size_t)N * HEADS;  // [N, 4]
    unsigned* maxb  = (unsigned*)(aldst + (size_t)N * HEADS);  // [N, 4] ordered-uint max
    float*    denom = (float*)(maxb + (size_t)N * HEADS);      // [N, 4]

    // Layer 1: [N,128] -> [N,64], relu
    run_layer(x,    W1, asrc1, adst1, b1, IN_C, HEADS, HID, 1,
              ei, N, E, hbuf, acc, alsrc, aldst, maxb, denom, xbuf, stream);
    // Layer 2: [N,64] -> [N,64], relu
    run_layer(xbuf, W2, asrc2, adst2, b2, D,    HEADS, HID, 1,
              ei, N, E, hbuf, acc, alsrc, aldst, maxb, denom, xbuf, stream);
    // Layer 3: [N,64] -> [N,32], single head, no relu, straight to d_out
    run_layer(xbuf, W3, asrc3, adst3, b3, D,    1, OUT_C, 0,
              ei, N, E, hbuf, acc, alsrc, aldst, maxb, denom, (float*)d_out, stream);
}
